// CUDALinearAttention_23398981829038
// MI455X (gfx1250) — matrix operand order repaired and checked by reading, not compiled, not run on hardware
//
#include <hip/hip_runtime.h>

typedef __attribute__((ext_vector_type(16))) _Float16 v16h;
typedef __attribute__((ext_vector_type(8)))  _Float16 v8h;
typedef __attribute__((ext_vector_type(8)))  float    v8f;
typedef __attribute__((ext_vector_type(4)))  unsigned u32x4;
typedef __attribute__((ext_vector_type(8)))  unsigned u32x8;

#define BB 4
#define T  4096
#define D  1024
#define H  16
#define HD 64
#define TK 32
#define TM 128

// ---- CDNA5 async global->LDS staging (ASYNCcnt-tracked, per-lane) ----
__device__ __forceinline__ void async_ld16(const _Float16* g, _Float16* l) {
  unsigned la = (unsigned)(size_t)l;                  // LDS byte offset = low 32 bits of flat addr
  unsigned long long ga = (unsigned long long)(size_t)g;
  asm volatile("global_load_async_to_lds_b128 %0, %1, off"
               :: "v"(la), "v"(ga) : "memory");
}
__device__ __forceinline__ void wait_async0() {
  asm volatile("s_wait_asynccnt 0" ::: "memory");
}

// ---- CDNA5 Tensor Data Mover: DMA a 2D tile (rows x 32 f16) into LDS ----
// D# per CDNA5 ISA: group0 {count,lds_addr,global_addr,type=2},
// group1 {data_size=2B, tensor_dim0=32, tensor_dim1=rows, tile_dim0=32, tile_dim1=rows,
// tensor_dim0_stride=strideElems}. 2-operand form => VADDR2/3 = NULL (<=2D tensor).
__device__ __forceinline__ void tdm_load_tile(unsigned lds_addr, const _Float16* gptr,
                                              unsigned rows, unsigned strideElems) {
  unsigned long long ga = (unsigned long long)(size_t)gptr;
  unsigned la   = (unsigned)__builtin_amdgcn_readfirstlane((int)lds_addr);
  unsigned galo = (unsigned)__builtin_amdgcn_readfirstlane((int)(unsigned)ga);
  unsigned gahi = (unsigned)__builtin_amdgcn_readfirstlane((int)(unsigned)(ga >> 32));
  u32x4 g0;
  g0[0] = 1u;                                          // count=1, no gather
  g0[1] = la;                                          // lds_addr
  g0[2] = galo;                                        // global_addr[31:0]
  g0[3] = 0x80000000u | (gahi & 0x01FFFFFFu);          // global_addr[56:32] | type=2
  u32x8 g1;
  g1[0] = 0x00010000u;                                 // wg_mask=0, data_size=1 (2B)
  g1[1] = 32u << 16;                                   // tensor_dim0[15:0]=32
  g1[2] = (rows & 0xFFFFu) << 16;                      // tensor_dim0[31:16]=0, tensor_dim1[15:0]=rows
  g1[3] = 32u << 16;                                   // tensor_dim1[31:16]=0, tile_dim0=32
  g1[4] = rows & 0xFFFFu;                              // tile_dim1=rows, tile_dim2=0
  g1[5] = strideElems;                                 // tensor_dim0_stride[31:0]
  g1[6] = 0u;                                          // stride hi, dim1_stride lo (unused, 2D)
  g1[7] = 0u;
  asm volatile("tensor_load_to_lds %0, %1" :: "s"(g0), "s"(g1) : "memory");
}

// ---- WMMA fragment gathers from LDS (f16, pair-packed as u32) ----
// A 16x32 f16: lane L holds row M=L%16; K interleave {0..7,16..23} (lanes 0-15)
// vs {8..15,24..31} (lanes 16-31); VGPR v holds K pair k0(v)= ((v&4)<<2)+((v&3)<<1)+half8
__device__ __forceinline__ v16h frag_a16(const unsigned* base, int row, int ustride, int half8) {
  v16h a;
#pragma unroll
  for (int v = 0; v < 8; ++v) {
    int k = ((v & 4) << 2) + ((v & 3) << 1) + half8;
    union { unsigned u; _Float16 h[2]; } c;
    c.u = base[row * ustride + (k >> 1)];
    a[2 * v]     = c.h[0];
    a[2 * v + 1] = c.h[1];
  }
  return a;
}

// B 32x16 f16: lane L holds col N=L%16; K interleave {0..7,16..23} (lanes 0-15)
// vs {8..15,24..31} (lanes 16-31), matching A. LDS tile stored [n][k] row-major.
__device__ __forceinline__ v16h frag_b16(const unsigned* base, int col, int ustride, int half16) {
  v16h b;
#pragma unroll
  for (int v = 0; v < 8; ++v) {
    int k = ((v & 4) << 2) + ((v & 3) << 1) + (half16 >> 1);
    union { unsigned u; _Float16 h[2]; } c;
    c.u = base[col * ustride + (k >> 1)];
    b[2 * v]     = c.h[0];
    b[2 * v + 1] = c.h[1];
  }
  return b;
}

__device__ __forceinline__ float phi_f(float x) {
  // elu(x)+1 : x>=0 -> x+1 ; x<0 -> exp(x)
  return x >= 0.f ? x + 1.f : __expf(x);
}

// =====================================================================
// One-shot f32 -> f16 conversion (x and the three weight matrices).
// =====================================================================
__global__ __launch_bounds__(256) void k_cvt(const float* __restrict__ src,
                                             _Float16* __restrict__ dst, int n) {
  int i = (blockIdx.x * 256 + threadIdx.x) * 8;
  if (i + 8 <= n) {
    v8h h;
#pragma unroll
    for (int j = 0; j < 8; ++j) h[j] = (_Float16)src[i + j];
    *(v8h*)(dst + i) = h;
  }
}

// =====================================================================
// Pass 1: K & V projection (WMMA, TDM-staged tiles) + phi/mask +
// partial kv = kf^T @ v and partial z = colsum(kf).
// =====================================================================
__global__ __launch_bounds__(256) void k_proj_kv(
    const _Float16* __restrict__ xh, const float* __restrict__ mask,
    const _Float16* __restrict__ wkh, const float* __restrict__ bk,
    const _Float16* __restrict__ wvh, const float* __restrict__ bv,
    float* __restrict__ kvPart, float* __restrict__ zPart)
{
  const int tile = blockIdx.x;   // 0..7  (512 t-rows each)
  const int h    = blockIdx.y;   // 0..15
  const int b    = blockIdx.z;   // 0..3
  const int tid  = threadIdx.x;
  const int lane = tid & 31;
  const int wave = tid >> 5;
  const int hi   = lane >> 4;    // lane half
  const int ln   = lane & 15;
  // scalar (SGPR) wave id -> uniform branch, since TDM ignores EXEC
  const int wave_s = __builtin_amdgcn_readfirstlane(tid) >> 5;

  __shared__ _Float16 Alds[2][TM * TK];    // x tile      [m=128][k=32], double buffered
  __shared__ _Float16 Blds[2][128 * TK];   // W tile      [n=128][k=32], double buffered
  __shared__ _Float16 KF[HD * TM];         // kf tile col-major [dk=64][m=128]
  __shared__ _Float16 VV[HD * TM];         // v  tile col-major [dv=64][m=128]
  __shared__ float mrow[TM];
  __shared__ float bias[128];

  if (tid < 64)       bias[tid] = bk[h * HD + tid];
  else if (tid < 128) bias[tid] = bv[h * HD + (tid - 64)];

  const int wm1 = wave >> 1, wn1 = wave & 1;  // GEMM1: 4x2 waves, 32x64 wave tile
  const int wm2 = wave >> 2, wn2 = wave & 3;  // GEMM2: 2x4 waves, 32x16 wave tile

  v8f ckv0 = {};
  v8f ckv1 = {};
  float zacc = 0.f;

  for (int c = 0; c < 4; ++c) {
    const int t0 = tile * 512 + c * TM;
    if (tid < TM) mrow[tid] = mask[(size_t)b * T + t0 + tid];

    // TDM stage of one (A, Bk, Bv) tile set into LDS buffer `buf` (wave 0 issues)
    auto stage = [&](int buf, int k0) {
      tdm_load_tile((unsigned)(size_t)&Alds[buf][0],
                    xh + ((size_t)b * T + t0) * D + k0, TM, D);
      tdm_load_tile((unsigned)(size_t)&Blds[buf][0],
                    wkh + (size_t)(h * HD) * D + k0, HD, D);
      tdm_load_tile((unsigned)(size_t)&Blds[buf][HD * TK],
                    wvh + (size_t)(h * HD) * D + k0, HD, D);
    };

    v8f acc[2][4];
#pragma unroll
    for (int i = 0; i < 2; ++i)
#pragma unroll
      for (int j = 0; j < 4; ++j) acc[i][j] = v8f{};

    int cur = 0;
    if (wave_s == 0) {
      stage(0, 0);
      __builtin_amdgcn_s_wait_tensorcnt(0);
    }
    __syncthreads();

    // ---- GEMM1: [128 x 1024] x [1024 x 128] with f16 WMMA ----
    for (int k0 = 0; k0 < D; k0 += TK) {
      if (wave_s == 0 && k0 + TK < D) stage(1 - cur, k0 + TK);  // DMA next tile
      const unsigned* Au = (const unsigned*)Alds[cur];
      const unsigned* Bu = (const unsigned*)Blds[cur];
      v16h af[2], bf[4];
#pragma unroll
      for (int i = 0; i < 2; ++i) af[i] = frag_a16(Au, wm1 * 32 + i * 16 + ln, 16, hi * 8);
#pragma unroll
      for (int j = 0; j < 4; ++j) bf[j] = frag_b16(Bu, wn1 * 64 + j * 16 + ln, 16, hi * 16);
#pragma unroll
      for (int i = 0; i < 2; ++i)
#pragma unroll
        for (int j = 0; j < 4; ++j)
          acc[i][j] = __builtin_amdgcn_wmma_f32_16x16x32_f16(
              false, af[i], false, bf[j], (short)0, acc[i][j], false, false);
      if (wave_s == 0) __builtin_amdgcn_s_wait_tensorcnt(0);
      __syncthreads();
      cur ^= 1;
    }

    // ---- epilogue: bias + phi + mask, store col-major f16 ----
#pragma unroll
    for (int i = 0; i < 2; ++i)
#pragma unroll
      for (int j = 0; j < 4; ++j) {
        int n = wn1 * 64 + j * 16 + ln;
        float bn = bias[n];
#pragma unroll
        for (int r = 0; r < 8; ++r) {
          int m = wm1 * 32 + i * 16 + hi * 8 + r;
          float val = acc[i][j][r] + bn;
          float mm = mrow[m];
          if (n < 64) KF[n * TM + m]        = (_Float16)(phi_f(val) * mm);
          else        VV[(n - 64) * TM + m] = (_Float16)(val * mm);
        }
      }
    __syncthreads();

    // z partial: column sums of kf
    if (tid < 64) {
      float s = 0.f;
      for (int m = 0; m < TM; ++m) s += (float)KF[tid * TM + m];
      zacc += s;
    }

    // ---- GEMM2: kv += kf^T @ v  (M=dk=64, N=dv=64, K=t=128) ----
    const unsigned* Ku = (const unsigned*)KF;  // [dk][t] row-major, 64 u32/row
    const unsigned* Vu = (const unsigned*)VV;  // [dv][t] row-major
#pragma unroll
    for (int ks = 0; ks < 4; ++ks) {
      const unsigned* Kp = Ku + ks * 16;  // advance K by 32 f16
      const unsigned* Vp = Vu + ks * 16;
      v16h a0 = frag_a16(Kp, wm2 * 32 + ln,      64, hi * 8);
      v16h a1 = frag_a16(Kp, wm2 * 32 + 16 + ln, 64, hi * 8);
      v16h b0 = frag_b16(Vp, wn2 * 16 + ln,      64, hi * 16);
      ckv0 = __builtin_amdgcn_wmma_f32_16x16x32_f16(false, a0, false, b0, (short)0, ckv0, false, false);
      ckv1 = __builtin_amdgcn_wmma_f32_16x16x32_f16(false, a1, false, b0, (short)0, ckv1, false, false);
    }
    __syncthreads();  // protect KF/VV/mrow before next chunk
  }

  // write deterministic partials
  const int slot = (b * H + h) * 8 + tile;
  float* kvp = kvPart + (size_t)slot * (HD * HD);
#pragma unroll
  for (int i = 0; i < 2; ++i) {
    v8f cc = (i == 0) ? ckv0 : ckv1;
#pragma unroll
    for (int r = 0; r < 8; ++r) {
      int m = wm2 * 32 + i * 16 + hi * 8 + r;
      int n = wn2 * 16 + ln;
      kvp[m * HD + n] = cc[r];
    }
  }
  if (tid < 64) zPart[(size_t)slot * HD + tid] = zacc;
}

// =====================================================================
// Pass 1b: deterministic reduction of the 8 partials per (b,h)
// =====================================================================
__global__ __launch_bounds__(256) void k_reduce(
    const float* __restrict__ kvPart, const float* __restrict__ zPart,
    float* __restrict__ kv, float* __restrict__ z)
{
  const int bh = blockIdx.x;
  for (int i = threadIdx.x; i < HD * HD; i += 256) {
    float s = 0.f;
#pragma unroll
    for (int t = 0; t < 8; ++t) s += kvPart[((size_t)bh * 8 + t) * (HD * HD) + i];
    kv[(size_t)bh * (HD * HD) + i] = s;
  }
  if (threadIdx.x < HD) {
    float s = 0.f;
#pragma unroll
    for (int t = 0; t < 8; ++t) s += zPart[((size_t)bh * 8 + t) * HD + threadIdx.x];
    z[bh * HD + threadIdx.x] = s;
  }
}

// =====================================================================
// Pass 2: Q projection (WMMA, async-DMA staged) + phi/mask,
// num = qf @ kv (WMMA), den = qf . z, out = num/den * mask
// =====================================================================
__global__ __launch_bounds__(256) void k_out(
    const _Float16* __restrict__ xh, const float* __restrict__ mask,
    const _Float16* __restrict__ wqh, const float* __restrict__ bq,
    const float* __restrict__ kv, const float* __restrict__ z,
    float* __restrict__ out)
{
  const int tile = blockIdx.x;   // 0..31 (128 t-rows)
  const int h    = blockIdx.y;
  const int b    = blockIdx.z;
  const int tid  = threadIdx.x;
  const int lane = tid & 31;
  const int wave = tid >> 5;
  const int hi   = lane >> 4;
  const int ln   = lane & 15;
  const int t0   = tile * TM;
  const int bh   = b * H + h;

  __shared__ _Float16 Alds[2][TM * TK];  // x tile [128][32], double buffered
  __shared__ _Float16 Blds[2][HD * TK];  // Wq tile [64][32], double buffered
  __shared__ _Float16 QF[TM * HD];       // qf row-major [m=128][k=64]
  __shared__ _Float16 KVB[HD * HD];      // kv transposed [dv=64][dk=64]
  __shared__ float zl[HD];
  __shared__ float den[TM];
  __shared__ float mrow[TM];
  __shared__ float bias[HD];

  if (tid < TM) mrow[tid] = mask[(size_t)b * T + t0 + tid];
  if (tid < HD) { bias[tid] = bq[h * HD + tid]; zl[tid] = z[bh * HD + tid]; }
  for (int e = tid; e < HD * HD; e += 256) {       // stage kv as B tile [n=dv][k=dk]
    int dk = e >> 6, dv = e & 63;
    KVB[dv * HD + dk] = (_Float16)kv[(size_t)bh * (HD * HD) + e];
  }

  auto stage = [&](int buf, int k0) {
#pragma unroll
    for (int i = 0; i < 2; ++i) {                 // A: 512 x 16B chunks
      int ch = tid + i * 256;
      int r = ch >> 2, sg = (ch & 3) * 8;
      async_ld16(xh + ((size_t)b * T + t0 + r) * D + k0 + sg,
                 &Alds[buf][r * TK + sg]);
    }
    {                                             // B: 256 x 16B chunks
      int ch = tid;
      int n = ch >> 2, sg = (ch & 3) * 8;
      async_ld16(wqh + (size_t)(h * HD + n) * D + k0 + sg,
                 &Blds[buf][n * TK + sg]);
    }
  };

  const int wm = wave >> 1, wn = wave & 1;  // 4x2 waves, 32x32 wave tiles
  v8f acc[2][2];
#pragma unroll
  for (int i = 0; i < 2; ++i)
#pragma unroll
    for (int j = 0; j < 2; ++j) acc[i][j] = v8f{};

  int cur = 0;
  stage(0, 0);
  wait_async0();
  __syncthreads();

  // ---- GEMM1: q projection [128 x 1024] x [1024 x 64] ----
  for (int k0 = 0; k0 < D; k0 += TK) {
    if (k0 + TK < D) stage(1 - cur, k0 + TK);
    v16h af[2], bf[2];
    af[0] = frag_a16((const unsigned*)Alds[cur], wm * 32 + ln,      16, hi * 8);
    af[1] = frag_a16((const unsigned*)Alds[cur], wm * 32 + 16 + ln, 16, hi * 8);
    bf[0] = frag_b16((const unsigned*)Blds[cur], wn * 32 + ln,      16, hi * 16);
    bf[1] = frag_b16((const unsigned*)Blds[cur], wn * 32 + 16 + ln, 16, hi * 16);
#pragma unroll
    for (int i = 0; i < 2; ++i)
#pragma unroll
      for (int j = 0; j < 2; ++j)
        acc[i][j] = __builtin_amdgcn_wmma_f32_16x16x32_f16(
            false, af[i], false, bf[j], (short)0, acc[i][j], false, false);
    wait_async0();
    __syncthreads();
    cur ^= 1;
  }

  // ---- epilogue: qf = phi(q + bias) * mask, row-major f16 ----
#pragma unroll
  for (int i = 0; i < 2; ++i)
#pragma unroll
    for (int j = 0; j < 2; ++j) {
      int n = wn * 32 + j * 16 + ln;
      float bn = bias[n];
#pragma unroll
      for (int r = 0; r < 8; ++r) {
        int m = wm * 32 + i * 16 + hi * 8 + r;
        QF[m * HD + n] = (_Float16)(phi_f(acc[i][j][r] + bn) * mrow[m]);
      }
    }
  __syncthreads();

  // ---- GEMM2: num = qf @ kv  (M=128, N=64, K=64) ----
  v8f o[2][2];
#pragma unroll
  for (int i = 0; i < 2; ++i)
#pragma unroll
    for (int j = 0; j < 2; ++j) o[i][j] = v8f{};
#pragma unroll
  for (int ks = 0; ks < 2; ++ks) {
    const unsigned* Qp = (const unsigned*)QF  + ks * 16;  // advance K by 32 f16
    const unsigned* Kp = (const unsigned*)KVB + ks * 16;
    v16h af[2], bf[2];
    af[0] = frag_a16(Qp, wm * 32 + ln,      32, hi * 8);
    af[1] = frag_a16(Qp, wm * 32 + 16 + ln, 32, hi * 8);
    bf[0] = frag_b16(Kp, wn * 32 + ln,      32, hi * 16);
    bf[1] = frag_b16(Kp, wn * 32 + 16 + ln, 32, hi * 16);
#pragma unroll
    for (int i = 0; i < 2; ++i)
#pragma unroll
      for (int j = 0; j < 2; ++j)
        o[i][j] = __builtin_amdgcn_wmma_f32_16x16x32_f16(
            false, af[i], false, bf[j], (short)0, o[i][j], false, false);
  }

  // ---- den = max(qf . z, 1e-6) per row ----
  if (tid < TM) {
    float s = 0.f;
    for (int k = 0; k < HD; ++k) s += (float)QF[tid * HD + k] * zl[k];
    den[tid] = fmaxf(s, 1e-6f);
  }
  __syncthreads();

  // ---- store out[b, t0+m, h*64+n] = num/den * mask ----
#pragma unroll
  for (int i = 0; i < 2; ++i)
#pragma unroll
    for (int j = 0; j < 2; ++j) {
#pragma unroll
      for (int r = 0; r < 8; ++r) {
        int m = wm * 32 + i * 16 + hi * 8 + r;
        int n = wn * 32 + j * 16 + ln;
        out[((size_t)b * T + t0 + m) * D + h * HD + n] = o[i][j][r] / den[m] * mrow[m];
      }
    }
}

extern "C" void kernel_launch(void* const* d_in, const int* in_sizes, int n_in,
                              void* d_out, int out_size, void* d_ws, size_t ws_size,
                              hipStream_t stream) {
  const float* x    = (const float*)d_in[0];
  const float* mask = (const float*)d_in[1];
  const float* Wq   = (const float*)d_in[2];
  const float* bq   = (const float*)d_in[3];
  const float* Wk   = (const float*)d_in[4];
  const float* bk   = (const float*)d_in[5];
  const float* Wv   = (const float*)d_in[6];
  const float* bv   = (const float*)d_in[7];
  float* out = (float*)d_out;

  // workspace layout:
  //   f16: xh[4*4096*1024], wqh/wkh/wvh[1024*1024]
  //   f32: kvPart[64][8][4096], zPart[64][8][64], kv[64][4096], z[64][64]
  const size_t NX = (size_t)BB * T * D;   // 16,777,216
  const size_t NW = (size_t)D * D;        //  1,048,576
  _Float16* xh  = (_Float16*)d_ws;
  _Float16* wqh = xh  + NX;
  _Float16* wkh = wqh + NW;
  _Float16* wvh = wkh + NW;
  float* kvPart = (float*)(wvh + NW);
  float* zPart  = kvPart + (size_t)64 * 8 * 4096;
  float* kvf    = zPart  + (size_t)64 * 8 * 64;
  float* zf     = kvf    + (size_t)64 * 4096;

  k_cvt<<<dim3((unsigned)(NX / 8 / 256)), 256, 0, stream>>>(x,  xh,  (int)NX);
  k_cvt<<<dim3((unsigned)(NW / 8 / 256)), 256, 0, stream>>>(Wq, wqh, (int)NW);
  k_cvt<<<dim3((unsigned)(NW / 8 / 256)), 256, 0, stream>>>(Wk, wkh, (int)NW);
  k_cvt<<<dim3((unsigned)(NW / 8 / 256)), 256, 0, stream>>>(Wv, wvh, (int)NW);

  k_proj_kv<<<dim3(8, 16, 4), 256, 0, stream>>>(xh, mask, wkh, bk, wvh, bv, kvPart, zPart);
  k_reduce<<<dim3(64), 256, 0, stream>>>(kvPart, zPart, kvf, zf);
  k_out<<<dim3(32, 16, 4), 256, 0, stream>>>(xh, mask, wqh, bq, kvf, zf, out);
}
